// Agent_68169720922419
// MI455X (gfx1250) — matrix operand order repaired and checked by reading, not compiled, not run on hardware
//
#include <hip/hip_runtime.h>
#include <hip/hip_bf16.h>
#include <stdint.h>

#define T_DIM 256
#define B_DIM 128
#define OBS   256
#define H_DIM 512
#define E_DIM 1024
#define DS    16
#define DC    4
#define DR    32

typedef __bf16 bf16_t;
typedef bf16_t v16bf __attribute__((ext_vector_type(16)));
typedef bf16_t v8bf  __attribute__((ext_vector_type(8)));
typedef float  v8f   __attribute__((ext_vector_type(8)));

__device__ __forceinline__ uint16_t f2bf(float f) {
  uint32_t u = __float_as_uint(f);
  uint32_t r = (u + 0x7FFFu + ((u >> 16) & 1u)) >> 16;
  return (uint16_t)r;
}
__device__ __forceinline__ float bf2f(uint16_t h) {
  return __uint_as_float(((uint32_t)h) << 16);
}
__device__ __forceinline__ v8f zero8() {
  v8f z;
#pragma unroll
  for (int i = 0; i < 8; ++i) z[i] = 0.f;
  return z;
}
__device__ __forceinline__ float siluf(float x) { return x / (1.0f + __expf(-x)); }
__device__ __forceinline__ float softplusf(float x) { return (x > 20.f) ? x : log1pf(__expf(x)); }

// A fragment: 16x32 bf16, row-major source with leading dim ld (in elements).
// ISA layout: lane<16 -> M=lane, K=[k0..k0+7],[k0+16..k0+23]; lane>=16 -> M=lane-16, K=[+8..+15],[+24..+31]
__device__ __forceinline__ v16bf ld_a16(const uint16_t* base, int ld, int k0) {
  int lane = threadIdx.x & 31;
  const uint16_t* p = base + (size_t)(lane & 15) * ld + k0 + ((lane >> 4) << 3);
  v8bf lo = *reinterpret_cast<const v8bf*>(p);
  v8bf hi = *reinterpret_cast<const v8bf*>(p + 16);
  return __builtin_shufflevector(lo, hi, 0,1,2,3,4,5,6,7,8,9,10,11,12,13,14,15);
}
// B fragment: 32x16 bf16, weight stored (N,K) row-major with leading dim ld.
// lane holds column N=lane&15, K=[k0+8h..+7],[k0+16+8h..+7] with h=(lane>=16)
__device__ __forceinline__ v16bf ld_b16(const uint16_t* base, int ld, int k0) {
  int lane = threadIdx.x & 31;
  const uint16_t* p = base + (size_t)(lane & 15) * ld + k0 + ((lane >> 4) << 3);
  v8bf lo = *reinterpret_cast<const v8bf*>(p);
  v8bf hi = *reinterpret_cast<const v8bf*>(p + 16);
  return __builtin_shufflevector(lo, hi, 0,1,2,3,4,5,6,7,8,9,10,11,12,13,14,15);
}
__device__ __forceinline__ v8f wmma_bf16(v16bf a, v16bf b, v8f c) {
  return __builtin_amdgcn_wmma_f32_16x16x32_bf16(false, a, false, b, (short)0, c, false, false);
}

// CDNA5 async global->LDS copy (16B per lane), tracked by ASYNCcnt.
// lds: wave-relative LDS byte offset (low 32 bits of the flat pointer to a __shared__ object,
// since the LDS aperture maps addr[31:0] directly to the LDS offset).
__device__ __forceinline__ void async_g2l_b128(uint32_t lds, uint64_t gaddr) {
  asm volatile("global_load_async_to_lds_b128 %0, %1, off" :: "v"(lds), "v"(gaddr) : "memory");
}
__device__ __forceinline__ void wait_asynccnt0() {
  asm volatile("s_wait_asynccnt 0x0" ::: "memory");
}

// ---------- utility kernels ----------
__global__ void cvt_f32_bf16(const float* __restrict__ src, uint16_t* __restrict__ dst, size_t n) {
  size_t i = (size_t)blockIdx.x * blockDim.x + threadIdx.x;
  size_t stride = (size_t)gridDim.x * blockDim.x;
  for (; i < n; i += stride) dst[i] = f2bf(src[i]);
}
__global__ void neg_exp_kernel(const float* __restrict__ src, float* __restrict__ dst, size_t n) {
  size_t i = (size_t)blockIdx.x * blockDim.x + threadIdx.x;
  size_t stride = (size_t)gridDim.x * blockDim.x;
  for (; i < n; i += stride) dst[i] = -__expf(src[i]);
}
__global__ void add_to_bf16(const float* __restrict__ a, const float* __restrict__ b,
                            uint16_t* __restrict__ dst, size_t n) {
  size_t i = (size_t)blockIdx.x * blockDim.x + threadIdx.x;
  size_t stride = (size_t)gridDim.x * blockDim.x;
  for (; i < n; i += stride) dst[i] = f2bf(a[i] + b[i]);
}

// ---------- bf16 WMMA GEMM: out = act(A(M,K) * B(N,K)^T + bias) ----------
// One wave computes a 16x64 strip (4 n-tiles) reusing its register-resident A fragments.
template <int K>
__global__ void gemm_bf16_t(const uint16_t* __restrict__ A, const uint16_t* __restrict__ Bw,
                            const float* __restrict__ bias, float* __restrict__ outF,
                            uint16_t* __restrict__ outB, int M, int N, int relu) {
  constexpr int KT = K / 32;
  int wid = threadIdx.x >> 5;
  int gw = blockIdx.x * (blockDim.x >> 5) + wid;
  int nquads = N >> 6;
  int mt = gw / nquads, nq = gw - mt * nquads;
  if (mt >= (M >> 4)) return;
  const uint16_t* abase = A + (size_t)mt * 16 * K;
  v16bf afr[KT];
#pragma unroll
  for (int k = 0; k < KT; ++k) afr[k] = ld_a16(abase, K, k * 32);
  int lane = threadIdx.x & 31;
  int n = lane & 15, ro = (lane >> 4) * 8;
#pragma unroll
  for (int q = 0; q < 4; ++q) {
    int nt = nq * 4 + q;
    const uint16_t* bbase = Bw + (size_t)nt * 16 * K;
    v8f acc = zero8();
#pragma unroll
    for (int k = 0; k < KT; ++k) acc = wmma_bf16(afr[k], ld_b16(bbase, K, k * 32), acc);
    int gn = nt * 16 + n;
    float bv = bias ? bias[gn] : 0.f;
#pragma unroll
    for (int v = 0; v < 8; ++v) {
      int gm = mt * 16 + ro + v;
      float val = acc[v] + bv;
      if (relu) val = fmaxf(val, 0.f);
      if (outF) outF[(size_t)gm * N + gn] = val;
      if (outB) outB[(size_t)gm * N + gn] = f2bf(val);
    }
  }
}

// ---------- persistent scan kernel: 8 WGs x 512 threads, each WG owns 16 batch rows ----------
__global__ void __launch_bounds__(512)
scan_kernel(const uint16_t* __restrict__ featsbf, const uint16_t* __restrict__ winbf,
            const uint16_t* __restrict__ wxprojbf, const uint16_t* __restrict__ wdtbf,
            const uint16_t* __restrict__ woutbf, const float* __restrict__ conv_state_in,
            const float* __restrict__ conv_w, const float* __restrict__ conv_b,
            const float* __restrict__ b_dt, const float* __restrict__ negA,
            const float* __restrict__ Dvec, const int* __restrict__ dones,
            float* __restrict__ ssmws, float* __restrict__ outseq) {
  __shared__ uint16_t sXZ[16][2 * E_DIM];    // xz (bf16); first half overwritten with y*silu(z)
  __shared__ uint16_t sCS[16][E_DIM][DC];    // conv state (bf16)
  __shared__ uint16_t sXC[16][E_DIM];        // silu(conv) output
  __shared__ uint16_t sDT[16][E_DIM];        // softplus(dt)
  __shared__ float    sXDB[16][DR + 2 * DS]; // xproj output (f32)
  __shared__ uint16_t sDTA[16][DR];          // bf16 copy of xdb[:, :DR] (A of dt GEMM)
  __shared__ uint16_t sFA[2][16][H_DIM];     // double-buffered feats tile (async-prefetched)

  const int tid = threadIdx.x;
  const int wid = tid >> 5;
  const int lane = tid & 31;
  const int r0 = blockIdx.x * 16;

  // init conv state from input (inputs are never mutated)
  for (int idx = tid; idx < 16 * E_DIM * DC; idx += 512) {
    int b = idx / (E_DIM * DC);
    int r = idx - b * (E_DIM * DC);
    sCS[b][r / DC][r % DC] = f2bf(conv_state_in[(size_t)(r0 + b) * E_DIM * DC + r]);
  }

  // async-prefetch feats tile for t=0 into buffer 0 (16KB: 512 thr x 2 x 16B)
  {
    const uint16_t* g0 = featsbf + ((size_t)0 * B_DIM + r0) * H_DIM;
    uint32_t lds0 = (uint32_t)(uintptr_t)&sFA[0][0][0];
    uint64_t ga = (uint64_t)(uintptr_t)g0;
    async_g2l_b128(lds0 + tid * 16, ga + tid * 16);
    async_g2l_b128(lds0 + tid * 16 + 8192, ga + tid * 16 + 8192);
  }
  wait_asynccnt0();
  __syncthreads();

  for (int t = 0; t < T_DIM; ++t) {
    // kick off async prefetch of next step's feats tile into the other buffer
    if (t + 1 < T_DIM) {
      const uint16_t* gn = featsbf + ((size_t)(t + 1) * B_DIM + r0) * H_DIM;
      uint32_t ldsn = (uint32_t)(uintptr_t)&sFA[(t + 1) & 1][0][0];
      uint64_t ga = (uint64_t)(uintptr_t)gn;
      async_g2l_b128(ldsn + tid * 16, ga + tid * 16);
      async_g2l_b128(ldsn + tid * 16 + 8192, ga + tid * 16 + 8192);
    }

    // ---- phase 1: xz = feats @ W_in^T  (16 x 2048, K=512), A from LDS ----
    {
      const uint16_t* abase = &sFA[t & 1][0][0];
      v16bf afr[16];
#pragma unroll
      for (int k = 0; k < 16; ++k) afr[k] = ld_a16(abase, H_DIM, k * 32);
      for (int nt = wid; nt < 128; nt += 16) {
        v8f acc = zero8();
        const uint16_t* bbase = winbf + (size_t)nt * 16 * H_DIM;
#pragma unroll
        for (int k = 0; k < 16; ++k)
          acc = wmma_bf16(afr[k], ld_b16(bbase, H_DIM, k * 32), acc);
        int n = lane & 15, ro = (lane >> 4) * 8;
#pragma unroll
        for (int v = 0; v < 8; ++v) sXZ[ro + v][nt * 16 + n] = f2bf(acc[v]);
      }
    }
    __syncthreads();

    // ---- phase 2: conv shift-in + silu ----
    for (int idx = tid; idx < 16 * E_DIM; idx += 512) {
      int b = idx >> 10, e = idx & (E_DIM - 1);
      uint16_t c1 = sCS[b][e][1], c2 = sCS[b][e][2], c3 = sCS[b][e][3];
      uint16_t xr = sXZ[b][e];
      sCS[b][e][0] = c1; sCS[b][e][1] = c2; sCS[b][e][2] = c3; sCS[b][e][3] = xr;
      float s = bf2f(c1) * conv_w[e * DC + 0] + bf2f(c2) * conv_w[e * DC + 1] +
                bf2f(c3) * conv_w[e * DC + 2] + bf2f(xr) * conv_w[e * DC + 3] + conv_b[e];
      sXC[b][e] = f2bf(siluf(s));
    }
    __syncthreads();

    // ---- phase 3: xdb = xc @ W_xproj^T  (16 x 64, K=1024) : waves 0..3 ----
    if (wid < 4) {
      int nt = wid;
      v8f acc = zero8();
      const uint16_t* bbase = wxprojbf + (size_t)nt * 16 * E_DIM;
      for (int k0 = 0; k0 < E_DIM; k0 += 32)
        acc = wmma_bf16(ld_a16(&sXC[0][0], E_DIM, k0), ld_b16(bbase, E_DIM, k0), acc);
      int n = lane & 15, ro = (lane >> 4) * 8;
#pragma unroll
      for (int v = 0; v < 8; ++v) {
        sXDB[ro + v][nt * 16 + n] = acc[v];
        if (nt < 2) sDTA[ro + v][nt * 16 + n] = f2bf(acc[v]);
      }
    }
    __syncthreads();

    // ---- phase 4: dt = softplus(xdb[:, :DR] @ W_dt^T + b_dt)  (16 x 1024, K=32) ----
    for (int nt = wid; nt < 64; nt += 16) {
      v8f acc = zero8();
      acc = wmma_bf16(ld_a16(&sDTA[0][0], DR, 0), ld_b16(wdtbf + (size_t)nt * 16 * DR, DR, 0), acc);
      int n = lane & 15, ro = (lane >> 4) * 8;
#pragma unroll
      for (int v = 0; v < 8; ++v) {
        int gn = nt * 16 + n;
        sDT[ro + v][gn] = f2bf(softplusf(acc[v] + b_dt[gn]));
      }
    }
    __syncthreads();

    // ---- phase 5: SSM recurrence + gating; write y*silu(z) into sXZ[:, :E] ----
    for (int idx = tid; idx < 16 * E_DIM; idx += 512) {
      int b = idx >> 10, e = idx & (E_DIM - 1);
      float mask = (dones[(size_t)t * B_DIM + r0 + b] != 0) ? 0.f : 1.f;
      float xcv = bf2f(sXC[b][e]);
      float dtv = bf2f(sDT[b][e]);
      float xdt = xcv * dtv;
      float y = Dvec[e] * xcv;
      size_t base = ((size_t)(r0 + b) * E_DIM + e) * DS;
#pragma unroll
      for (int n = 0; n < DS; ++n) {
        float dA = __expf(dtv * negA[e * DS + n]);
        float Bv = sXDB[b][DR + n];
        float Cv = sXDB[b][DR + DS + n];
        float s = ssmws[base + n] * dA + xdt * Bv;
        y += s * Cv;                 // y uses pre-mask state (reference order)
        ssmws[base + n] = s * mask;  // carry is masked
      }
      if (mask == 0.f) { sCS[b][e][0] = 0; sCS[b][e][1] = 0; sCS[b][e][2] = 0; sCS[b][e][3] = 0; }
      float z = bf2f(sXZ[b][E_DIM + e]);
      sXZ[b][e] = f2bf(y * siluf(z));
    }
    __syncthreads();

    // ---- phase 6: out = (y*silu(z)) @ W_out^T  (16 x 512, K=1024) ----
    for (int nt = wid * 2; nt < wid * 2 + 2; ++nt) {
      v8f acc = zero8();
      const uint16_t* bbase = woutbf + (size_t)nt * 16 * E_DIM;
      for (int k0 = 0; k0 < E_DIM; k0 += 32)
        acc = wmma_bf16(ld_a16(&sXZ[0][0], 2 * E_DIM, k0), ld_b16(bbase, E_DIM, k0), acc);
      int n = lane & 15, ro = (lane >> 4) * 8;
#pragma unroll
      for (int v = 0; v < 8; ++v)
        outseq[((size_t)t * B_DIM + r0 + ro + v) * H_DIM + nt * 16 + n] = acc[v];
    }

    // drain this step's async prefetch before next iteration reads/overwrites buffers
    wait_asynccnt0();
    __syncthreads();
  }
}

// ---------- final: h2 = h1 @ W2^T + b2 -> LayerNorm -> out ----------
__global__ void __launch_bounds__(512)
final_kernel(const uint16_t* __restrict__ h1, const uint16_t* __restrict__ W2,
             const float* __restrict__ b2, const float* __restrict__ gamma,
             const float* __restrict__ beta, float* __restrict__ out) {
  __shared__ float sH[16][H_DIM];
  int wid = threadIdx.x >> 5;
  int lane = threadIdx.x & 31;
  int mt = blockIdx.x;
  const uint16_t* abase = h1 + (size_t)mt * 16 * H_DIM;
  for (int nt = wid * 2; nt < wid * 2 + 2; ++nt) {
    v8f acc = zero8();
    const uint16_t* bbase = W2 + (size_t)nt * 16 * H_DIM;
    for (int k0 = 0; k0 < H_DIM; k0 += 32)
      acc = wmma_bf16(ld_a16(abase, H_DIM, k0), ld_b16(bbase, H_DIM, k0), acc);
    int n = lane & 15, ro = (lane >> 4) * 8;
#pragma unroll
    for (int v = 0; v < 8; ++v) sH[ro + v][nt * 16 + n] = acc[v] + b2[nt * 16 + n];
  }
  __syncthreads();
  // LayerNorm: wave `wid` owns row `wid`
  float s = 0.f, s2 = 0.f;
  for (int j = lane; j < H_DIM; j += 32) { float v = sH[wid][j]; s += v; s2 += v * v; }
#pragma unroll
  for (int off = 16; off > 0; off >>= 1) { s += __shfl_xor(s, off, 32); s2 += __shfl_xor(s2, off, 32); }
  float mu = s / H_DIM;
  float var = s2 / H_DIM - mu * mu;
  float rstd = rsqrtf(var + 1e-5f);
  size_t rowbase = ((size_t)mt * 16 + wid) * H_DIM;
  for (int j = lane; j < H_DIM; j += 32)
    out[rowbase + j] = (sH[wid][j] - mu) * rstd * gamma[j] + beta[j];
}

extern "C" void kernel_launch(void* const* d_in, const int* in_sizes, int n_in,
                              void* d_out, int out_size, void* d_ws, size_t ws_size,
                              hipStream_t stream) {
  (void)in_sizes; (void)n_in; (void)out_size; (void)ws_size;
  const float* x          = (const float*)d_in[0];
  const int*   dones      = (const int*)d_in[1];
  const float* conv_state = (const float*)d_in[2];
  const float* ssm_state  = (const float*)d_in[3];
  const float* W_enc      = (const float*)d_in[4];
  const float* b_enc      = (const float*)d_in[5];
  const float* W_in       = (const float*)d_in[6];
  const float* conv_w     = (const float*)d_in[7];
  const float* conv_b     = (const float*)d_in[8];
  const float* W_xproj    = (const float*)d_in[9];
  const float* W_dt       = (const float*)d_in[10];
  const float* b_dt       = (const float*)d_in[11];
  const float* A_log      = (const float*)d_in[12];
  const float* Dvec       = (const float*)d_in[13];
  const float* W_out      = (const float*)d_in[14];
  const float* W1         = (const float*)d_in[15];
  const float* b1         = (const float*)d_in[16];
  const float* W2         = (const float*)d_in[17];
  const float* b2         = (const float*)d_in[18];
  const float* gamma      = (const float*)d_in[19];
  const float* beta       = (const float*)d_in[20];
  float* out = (float*)d_out;

  const size_t TB = (size_t)T_DIM * B_DIM;
  size_t off = 0;
  auto alloc = [&](size_t bytes) -> void* {
    void* p = (char*)d_ws + off;
    off += (bytes + 255) & ~(size_t)255;
    return p;
  };
  uint16_t* xbf      = (uint16_t*)alloc(TB * OBS * 2);
  uint16_t* wencbf   = (uint16_t*)alloc((size_t)H_DIM * OBS * 2);
  float*    featsf   = (float*)alloc(TB * H_DIM * 4);
  uint16_t* featsbf  = (uint16_t*)alloc(TB * H_DIM * 2);
  uint16_t* winbf    = (uint16_t*)alloc((size_t)2 * E_DIM * H_DIM * 2);
  uint16_t* wxprojbf = (uint16_t*)alloc((size_t)(DR + 2 * DS) * E_DIM * 2);
  uint16_t* wdtbf    = (uint16_t*)alloc((size_t)E_DIM * DR * 2);
  uint16_t* woutbf   = (uint16_t*)alloc((size_t)H_DIM * E_DIM * 2);
  uint16_t* w1bf     = (uint16_t*)alloc((size_t)H_DIM * H_DIM * 2);
  uint16_t* w2bf     = (uint16_t*)alloc((size_t)H_DIM * H_DIM * 2);
  float*    negA     = (float*)alloc((size_t)E_DIM * DS * 4);
  float*    ssmws    = (float*)alloc((size_t)B_DIM * E_DIM * DS * 4);
  float*    outseq   = (float*)alloc(TB * H_DIM * 4);
  uint16_t* hsumbf   = (uint16_t*)alloc(TB * H_DIM * 2);
  uint16_t* h1bf     = (uint16_t*)alloc(TB * H_DIM * 2);

  // convert inputs/weights to bf16
  cvt_f32_bf16<<<4096, 256, 0, stream>>>(x, xbf, TB * OBS);
  cvt_f32_bf16<<<512, 256, 0, stream>>>(W_enc, wencbf, (size_t)H_DIM * OBS);
  cvt_f32_bf16<<<2048, 256, 0, stream>>>(W_in, winbf, (size_t)2 * E_DIM * H_DIM);
  cvt_f32_bf16<<<256, 256, 0, stream>>>(W_xproj, wxprojbf, (size_t)(DR + 2 * DS) * E_DIM);
  cvt_f32_bf16<<<128, 256, 0, stream>>>(W_dt, wdtbf, (size_t)E_DIM * DR);
  cvt_f32_bf16<<<1024, 256, 0, stream>>>(W_out, woutbf, (size_t)H_DIM * E_DIM);
  cvt_f32_bf16<<<512, 256, 0, stream>>>(W1, w1bf, (size_t)H_DIM * H_DIM);
  cvt_f32_bf16<<<512, 256, 0, stream>>>(W2, w2bf, (size_t)H_DIM * H_DIM);
  neg_exp_kernel<<<64, 256, 0, stream>>>(A_log, negA, (size_t)E_DIM * DS);
  // working copy of SSM state (inputs must not be mutated)
  hipMemcpyAsync(ssmws, ssm_state, (size_t)B_DIM * E_DIM * DS * 4, hipMemcpyDeviceToDevice, stream);

  // feats = relu(x @ W_enc^T + b_enc): M=32768, N=512, K=256
  gemm_bf16_t<OBS><<<2048, 256, 0, stream>>>(xbf, wencbf, b_enc, featsf, featsbf,
                                             (int)TB, H_DIM, 1);

  // persistent recurrent scan: 8 WGs, each owns 16 batch rows across all 256 steps
  scan_kernel<<<B_DIM / 16, 512, 0, stream>>>(featsbf, winbf, wxprojbf, wdtbf, woutbf,
                                              conv_state, conv_w, conv_b, b_dt, negA, Dvec,
                                              dones, ssmws, outseq);

  // h = out_seq + feats (residual), then MLP head
  add_to_bf16<<<8192, 256, 0, stream>>>(outseq, featsf, hsumbf, TB * H_DIM);
  gemm_bf16_t<H_DIM><<<2048, 256, 0, stream>>>(hsumbf, w1bf, b1, nullptr, h1bf,
                                               (int)TB, H_DIM, 1);
  final_kernel<<<(int)(TB / 16), 512, 0, stream>>>(h1bf, w2bf, b2, gamma, beta, out);
}
